// MultiheadTaylorAttention_28269474742886
// MI455X (gfx1250) — matrix operand order repaired and checked by reading, not compiled, not run on hardware
//
#include <hip/hip_runtime.h>
#include <hip/hip_bf16.h>

#define L_SEQ 2048
#define BATCH 2
#define EMB   1024
#define NH    16
#define HD    64
#define ROWS  (L_SEQ * BATCH)   // 4096
#define BH    (BATCH * NH)      // 32

typedef _Float16 half_t;
typedef __attribute__((ext_vector_type(16))) _Float16 v16h;
typedef __attribute__((ext_vector_type(8)))  _Float16 v8h;
typedef __attribute__((ext_vector_type(8)))  float    v8f;
typedef __attribute__((ext_vector_type(4)))  float    v4f;

// ---------------------------------------------------------------------------
// Fragment helpers (wave32, CDNA5 16-bit WMMA layouts, CDNA5 ISA §7.12.2)
// A 16x32 f16: lane g=0 holds K 0-7 then 16-23; g=1 holds K 8-15 then 24-31.
// B 32x16 f16: lane n holds K 0-7 then 16-23 (g=0); K 8-15 then 24-31 (g=1).
// ---------------------------------------------------------------------------
__device__ inline v16h frag_join(v8h lo, v8h hi) {
  v16h r;
#pragma unroll
  for (int i = 0; i < 8; ++i) { r[i] = lo[i]; r[i + 8] = hi[i]; }
  return r;
}

// A-fragment from an f32 row-major matrix (convert inline to f16)
__device__ inline v16h load_a_frag_f32(const float* __restrict__ base, int ld,
                                       int row0, int k0) {
  const int lane = threadIdx.x & 31;
  const int m = lane & 15, g = lane >> 4;
  const float* p = base + (size_t)(row0 + m) * ld + k0 + g * 8;
  v4f a0 = *(const v4f*)(p + 0);
  v4f a1 = *(const v4f*)(p + 4);
  v4f a2 = *(const v4f*)(p + 16);
  v4f a3 = *(const v4f*)(p + 20);
  v16h r;
#pragma unroll
  for (int i = 0; i < 4; ++i) {
    r[i]      = (_Float16)a0[i];
    r[i + 4]  = (_Float16)a1[i];
    r[i + 8]  = (_Float16)a2[i];
    r[i + 12] = (_Float16)a3[i];
  }
  return r;
}

// A-fragment from an f16 row-major matrix
__device__ inline v16h load_a_frag_f16(const half_t* __restrict__ base, int ld,
                                       int row0, int k0) {
  const int lane = threadIdx.x & 31;
  const int m = lane & 15, g = lane >> 4;
  const half_t* p = base + (size_t)(row0 + m) * ld + k0 + g * 8;
  v8h lo = *(const v8h*)(p);
  v8h hi = *(const v8h*)(p + 16);
  return frag_join(lo, hi);
}

// B-fragment from an f16 N-major (pre-transposed) matrix: base[n][k]
__device__ inline v16h load_b_frag_f16(const half_t* __restrict__ base, int ld,
                                       int col0, int k0) {
  const int lane = threadIdx.x & 31;
  const int n = lane & 15, g = lane >> 4;
  const half_t* p = base + (size_t)(col0 + n) * ld + k0 + g * 8;
  v8h lo = *(const v8h*)(p);
  v8h hi = *(const v8h*)(p + 16);
  return frag_join(lo, hi);
}

__device__ inline v8f wmma_f16(v16h a, v16h b, v8f c) {
  return __builtin_amdgcn_wmma_f32_16x16x32_f16(false, a, false, b,
                                                (short)0, c, false, false);
}

// ---------------------------------------------------------------------------
// 1) Weight transpose + f32->f16: Wt[n][k] = W[k][n]
// ---------------------------------------------------------------------------
__global__ __launch_bounds__(256) void wt_kernel(const float* __restrict__ W,
                                                 half_t* __restrict__ Wt) {
  int idx = blockIdx.x * 256 + threadIdx.x;    // coalesced read over n
  int k = idx >> 10, n = idx & (EMB - 1);
  Wt[(size_t)n * EMB + k] = (half_t)W[idx];
}

// ---------------------------------------------------------------------------
// 2) Projection GEMM: C(4096x1024) = X(f32) @ W, +bias, *scale, store f16.
//    mode 0: Q -> Qh[bh][l][d] (d contiguous)
//    mode 1: K/V -> Xt[bh][d][t] (t contiguous, transposed for phase 2)
// ---------------------------------------------------------------------------
__global__ __launch_bounds__(32) void proj_kernel(
    const float* __restrict__ X, const half_t* __restrict__ Wt,
    const float* __restrict__ bias, half_t* __restrict__ out,
    float scale, int mode) {
  const int row0 = blockIdx.x * 64;
  const int col0 = blockIdx.y * 64;
  v8f c[4][4] = {};
  for (int k0 = 0; k0 < EMB; k0 += 32) {
    if (k0 + 32 < EMB)
      __builtin_prefetch(X + (size_t)(row0 + (threadIdx.x & 15)) * EMB + k0 + 32, 0, 3);
    v16h a[4], b[4];
#pragma unroll
    for (int i = 0; i < 4; ++i) a[i] = load_a_frag_f32(X, EMB, row0 + 16 * i, k0);
#pragma unroll
    for (int j = 0; j < 4; ++j) b[j] = load_b_frag_f16(Wt, EMB, col0 + 16 * j, k0);
#pragma unroll
    for (int i = 0; i < 4; ++i)
#pragma unroll
      for (int j = 0; j < 4; ++j) c[i][j] = wmma_f16(a[i], b[j], c[i][j]);
  }
  const int lane = threadIdx.x & 31;
  const int nl = lane & 15, g = lane >> 4;
#pragma unroll
  for (int j = 0; j < 4; ++j) {
    const int n = col0 + 16 * j + nl;
    const float bv = bias[n];
    const int h = n >> 6, d = n & (HD - 1);
#pragma unroll
    for (int i = 0; i < 4; ++i) {
#pragma unroll
      for (int r = 0; r < 8; ++r) {
        const int m = row0 + 16 * i + g * 8 + r;      // global row = t*B + b
        const int t = m >> 1, bb = m & 1;
        const float val = (c[i][j][r] + bv) * scale;
        size_t off;
        if (mode == 0) off = ((size_t)(bb * NH + h) * L_SEQ + t) * HD + d;
        else           off = ((size_t)(bb * NH + h) * HD + d) * L_SEQ + t;
        out[off] = (half_t)val;
      }
    }
  }
}

// ---------------------------------------------------------------------------
// 3) Column sums of a [bh][d][s] f16 tensor -> S[bh][d] f32
// ---------------------------------------------------------------------------
__global__ __launch_bounds__(64) void sum_kernel(const half_t* __restrict__ T,
                                                 float* __restrict__ S) {
  const int bh = blockIdx.x, d = threadIdx.x;
  const half_t* p = T + ((size_t)bh * HD + d) * L_SEQ;
  float s = 0.f;
  for (int i = 0; i < L_SEQ; i += 8) {
    v8h v = *(const v8h*)(p + i);
#pragma unroll
    for (int j = 0; j < 8; ++j) s += (float)v[j];
  }
  S[bh * HD + d] = s;
}

// ---------------------------------------------------------------------------
// 4) KV[bh] = K_h^T @ V_h  (64x64, K-loop over S=2048), stored column-major
//    f16 (KVc[bh][d2][d1]) so it can be used directly as a B operand.
// ---------------------------------------------------------------------------
__global__ __launch_bounds__(32) void kv_kernel(const half_t* __restrict__ Kt,
                                                const half_t* __restrict__ Vt,
                                                half_t* __restrict__ KVc) {
  const int bh = blockIdx.x;
  const int m0 = blockIdx.y * 16;   // d1 tile
  const int n0 = blockIdx.z * 16;   // d2 tile
  const half_t* Kb = Kt + (size_t)bh * HD * L_SEQ;
  const half_t* Vb = Vt + (size_t)bh * HD * L_SEQ;
  v8f c = {};
  for (int k0 = 0; k0 < L_SEQ; k0 += 32) {
    v16h a = load_a_frag_f16(Kb, L_SEQ, m0, k0);   // A[d1][s]
    v16h b = load_b_frag_f16(Vb, L_SEQ, n0, k0);   // B[s][d2] via Vt[d2][s]
    c = wmma_f16(a, b, c);
  }
  const int lane = threadIdx.x & 31;
  const int nl = lane & 15, g = lane >> 4;
  half_t* outb = KVc + (size_t)bh * HD * HD;
#pragma unroll
  for (int r = 0; r < 8; ++r) {
    const int m = m0 + g * 8 + r;   // d1
    const int n = n0 + nl;          // d2
    outb[(size_t)n * HD + m] = (half_t)c[r];
  }
}

// ---------------------------------------------------------------------------
// 5) denom[bh][l] = S + q_l . ksum
// ---------------------------------------------------------------------------
__global__ __launch_bounds__(256) void denom_kernel(
    const half_t* __restrict__ Qh, const float* __restrict__ ksum,
    float* __restrict__ denom) {
  const int bh = blockIdx.x;
  const int l = blockIdx.y * 256 + threadIdx.x;
  const half_t* q = Qh + ((size_t)bh * L_SEQ + l) * HD;
  const float* ks = ksum + bh * HD;
  float s = (float)L_SEQ;
#pragma unroll 8
  for (int d = 0; d < HD; ++d) s += (float)q[d] * ks[d];
  denom[bh * L_SEQ + l] = s;
}

// ---------------------------------------------------------------------------
// 6) out = (vsum + Q @ KV) / denom, scattered back to (L, B, E) f32
// ---------------------------------------------------------------------------
__global__ __launch_bounds__(32) void out_kernel(
    const half_t* __restrict__ Qh, const half_t* __restrict__ KVc,
    const float* __restrict__ vsum, const float* __restrict__ denom,
    float* __restrict__ out) {
  const int bh = blockIdx.x;
  const int l0 = blockIdx.y * 64;
  const int b = bh >> 4, h = bh & (NH - 1);
  const half_t* Qb = Qh + (size_t)bh * L_SEQ * HD;
  const half_t* Kb = KVc + (size_t)bh * HD * HD;
  v8f c[4][4] = {};
#pragma unroll
  for (int k0 = 0; k0 < HD; k0 += 32) {
    v16h a[4], bf[4];
#pragma unroll
    for (int i = 0; i < 4; ++i) a[i] = load_a_frag_f16(Qb, HD, l0 + 16 * i, k0);
#pragma unroll
    for (int j = 0; j < 4; ++j) bf[j] = load_b_frag_f16(Kb, HD, 16 * j, k0);
#pragma unroll
    for (int i = 0; i < 4; ++i)
#pragma unroll
      for (int j = 0; j < 4; ++j) c[i][j] = wmma_f16(a[i], bf[j], c[i][j]);
  }
  const int lane = threadIdx.x & 31;
  const int nl = lane & 15, g = lane >> 4;
  const float* vs = vsum + bh * HD;
#pragma unroll
  for (int j = 0; j < 4; ++j) {
    const int d = 16 * j + nl;
    const float vsd = vs[d];
#pragma unroll
    for (int i = 0; i < 4; ++i) {
#pragma unroll
      for (int r = 0; r < 8; ++r) {
        const int lm = l0 + 16 * i + g * 8 + r;
        const float den = denom[bh * L_SEQ + lm];
        const float val = (c[i][j][r] + vsd) / den;
        out[((size_t)lm * BATCH + b) * EMB + h * HD + d] = val;
      }
    }
  }
}

// ---------------------------------------------------------------------------
extern "C" void kernel_launch(void* const* d_in, const int* in_sizes, int n_in,
                              void* d_out, int out_size, void* d_ws, size_t ws_size,
                              hipStream_t stream) {
  const float* q  = (const float*)d_in[0];
  const float* k  = (const float*)d_in[1];
  const float* v  = (const float*)d_in[2];
  const float* Wq = (const float*)d_in[3];
  const float* bq = (const float*)d_in[4];
  const float* Wk = (const float*)d_in[5];
  const float* bk = (const float*)d_in[6];
  const float* Wv = (const float*)d_in[7];
  const float* bv = (const float*)d_in[8];
  float* out = (float*)d_out;

  char* ws = (char*)d_ws;
  const size_t MB = 1024 * 1024;
  half_t* WtQ = (half_t*)(ws + 0 * MB);          // 2 MB each
  half_t* WtK = (half_t*)(ws + 2 * MB);
  half_t* WtV = (half_t*)(ws + 4 * MB);
  half_t* QhP = (half_t*)(ws + 6 * MB);          // 8 MB  [bh][l][d]
  half_t* KtP = (half_t*)(ws + 14 * MB);         // 8 MB  [bh][d][s]
  half_t* VtP = (half_t*)(ws + 22 * MB);         // 8 MB  [bh][d][s]
  half_t* KVc = (half_t*)(ws + 30 * MB);         // 256 KB [bh][d2][d1]
  float*  ksum = (float*)(ws + 30 * MB + 512 * 1024);   // 8 KB
  float*  vsum = ksum + BH * HD;                         // 8 KB
  float*  den  = vsum + BH * HD;                         // 256 KB

  // 1) transpose + convert weights
  wt_kernel<<<dim3(EMB * EMB / 256), dim3(256), 0, stream>>>(Wq, WtQ);
  wt_kernel<<<dim3(EMB * EMB / 256), dim3(256), 0, stream>>>(Wk, WtK);
  wt_kernel<<<dim3(EMB * EMB / 256), dim3(256), 0, stream>>>(Wv, WtV);

  // 2) projections (WMMA); q pre-scaled by E^-0.5 = 1/32
  dim3 pg(ROWS / 64, EMB / 64);
  proj_kernel<<<pg, dim3(32), 0, stream>>>(q, WtQ, bq, QhP, 0.03125f, 0);
  proj_kernel<<<pg, dim3(32), 0, stream>>>(k, WtK, bk, KtP, 1.0f, 1);
  proj_kernel<<<pg, dim3(32), 0, stream>>>(v, WtV, bv, VtP, 1.0f, 1);

  // 3) k/v column sums
  sum_kernel<<<dim3(BH), dim3(64), 0, stream>>>(KtP, ksum);
  sum_kernel<<<dim3(BH), dim3(64), 0, stream>>>(VtP, vsum);

  // 4) KV = K^T V per head (WMMA)
  kv_kernel<<<dim3(BH, HD / 16, HD / 16), dim3(32), 0, stream>>>(KtP, VtP, KVc);

  // 5) row denominators
  denom_kernel<<<dim3(BH, L_SEQ / 256), dim3(256), 0, stream>>>(QhP, ksum, den);

  // 6) output GEMM + normalization (WMMA)
  out_kernel<<<dim3(BH, L_SEQ / 64), dim3(32), 0, stream>>>(QhP, KVc, vsum, den, out);
}
